// QJLWithOrthogonal_42700564857503
// MI455X (gfx1250) — matrix operand order repaired and checked by reading, not compiled, not run on hardware
//
#include <hip/hip_runtime.h>
#include <math.h>

// Problem constants (from the reference)
#define DIM        128      // feature dim (K of the GEMM)
#define NPROJ      256      // number of projections (N of the GEMM)
#define ROWS_PB    128      // rows per block (8 waves x 16 rows)
#define THREADS    256      // 8 wave32s

typedef __attribute__((ext_vector_type(16))) __bf16        v16bf;
typedef __attribute__((ext_vector_type(8)))  float         v8f;
typedef __attribute__((ext_vector_type(4)))  float         v4f;
typedef __attribute__((ext_vector_type(4)))  unsigned int  v4u;
typedef __attribute__((ext_vector_type(2)))  unsigned int  v2u;

union Frag {                 // one WMMA 16x32 bf16 operand per lane
    v16bf        v;
    v4u          q[2];
    unsigned int u[8];
};

// fp32 -> bf16 (round-to-nearest-even), two at a time packed into a DWORD
__device__ __forceinline__ unsigned int pack_bf16(float a, float b) {
    unsigned int ua = __float_as_uint(a);
    unsigned int ub = __float_as_uint(b);
    ua += 0x7fffu + ((ua >> 16) & 1u);
    ub += 0x7fffu + ((ub >> 16) & 1u);
    return (ua >> 16) | (ub & 0xffff0000u);
}

__global__ __launch_bounds__(THREADS)
void qjl_sign_proj_kernel(const float* __restrict__ query,
                          const float* __restrict__ key,
                          const float* __restrict__ S,
                          float* __restrict__ out,
                          int rows) {
    // --- LDS: S converted to bf16, row-major [NPROJ][DIM] ---------------
    __shared__ __align__(16) unsigned short sS[NPROJ * DIM];   // 64 KB

    const int tid     = threadIdx.x;
    const int rowBase = blockIdx.x * ROWS_PB;

    // Stage S: fp32 -> bf16, fully coalesced (float4 in, uint2 out)
    {
        const v4f* s4 = (const v4f*)S;
        v2u*       d2 = (v2u*)sS;
        #pragma unroll 4
        for (int i = tid; i < (NPROJ * DIM) / 4; i += THREADS) {
            v4f f = s4[i];
            v2u p;
            p[0] = pack_bf16(f[0], f[1]);
            p[1] = pack_bf16(f[2], f[3]);
            d2[i] = p;
        }
    }
    __syncthreads();

    // --- Per-wave 16-row tile -------------------------------------------
    const int wave = tid >> 5;          // 0..7
    const int lane = tid & 31;
    const int r    = lane & 15;         // row-in-tile (A) / N-in-chunk (B)
    const int hi   = lane >> 4;         // half-wave selector

    const int myRow = rowBase + wave * 16 + r;

    // Load A fragments (q and k) from global, convert fp32 -> bf16.
    // A as packed here (k order shared with B): lanes 0-15 hold K = kb+[0..15],
    // lanes 16-31 hold K = kb+[16..31], row M = lane&15.
    Frag aQ[4], aK[4];
    {
        const float* rq = query + (size_t)myRow * DIM;
        const float* rk = key   + (size_t)myRow * DIM;
        #pragma unroll
        for (int ks = 0; ks < 4; ++ks) {
            const int c0 = ks * 32 + hi * 16;    // first 8-wide K chunk
            #pragma unroll
            for (int h = 0; h < 2; ++h) {        // h=0 -> +0, h=1 -> +8
                const int c = c0 + h * 8;
                v4f f0 = *(const v4f*)(rq + c);
                v4f f1 = *(const v4f*)(rq + c + 4);
                aQ[ks].u[4 * h + 0] = pack_bf16(f0[0], f0[1]);
                aQ[ks].u[4 * h + 1] = pack_bf16(f0[2], f0[3]);
                aQ[ks].u[4 * h + 2] = pack_bf16(f1[0], f1[1]);
                aQ[ks].u[4 * h + 3] = pack_bf16(f1[2], f1[3]);
                v4f g0 = *(const v4f*)(rk + c);
                v4f g1 = *(const v4f*)(rk + c + 4);
                aK[ks].u[4 * h + 0] = pack_bf16(g0[0], g0[1]);
                aK[ks].u[4 * h + 1] = pack_bf16(g0[2], g0[3]);
                aK[ks].u[4 * h + 2] = pack_bf16(g1[0], g1[1]);
                aK[ks].u[4 * h + 3] = pack_bf16(g1[2], g1[3]);
            }
        }
    }

    // Running per-row sums: VGPR j holds row (hi*8 + j) of this tile,
    // lane axis = projection index within the current 16-wide N chunk.
    float racc[8];
    #pragma unroll
    for (int j = 0; j < 8; ++j) racc[j] = 0.0f;

    // Loop over 16 projection chunks of 16; B comes from LDS.
    // B as loaded here (k order shared with A): lane holds N = nb + (lane&15),
    // K = kb + 16*hi + [0..15] -> 32 contiguous bytes in sS.
    for (int nb = 0; nb < NPROJ; nb += 16) {
        v8f cQ = {};
        v8f cK = {};
        const unsigned short* bp = sS + (nb + r) * DIM + hi * 16;
        #pragma unroll
        for (int ks = 0; ks < 4; ++ks) {
            Frag b;
            const v4u* p = (const v4u*)(bp + ks * 32);
            b.q[0] = p[0];
            b.q[1] = p[1];
            cQ = __builtin_amdgcn_wmma_f32_16x16x32_bf16(
                    false, aQ[ks].v, false, b.v, (short)0, cQ, false, false);
            cK = __builtin_amdgcn_wmma_f32_16x16x32_bf16(
                    false, aK[ks].v, false, b.v, (short)0, cK, false, false);
        }
        // qp * sign(kp) as a sign-bit transfer: flip qp's sign bit with
        // kp's sign bit (v_and + v_xor + v_add, all dual-issueable; avoids
        // the 2x v_cmp + v_cndmask chain and its scalar mask pressure).
        #pragma unroll
        for (int j = 0; j < 8; ++j) {
            unsigned int sgn = __float_as_uint(cK[j]) & 0x80000000u;
            racc[j] += __uint_as_float(__float_as_uint(cQ[j]) ^ sgn);
        }
    }

    // Reduce over the 16-lane half (projection/lane axis).
    #pragma unroll
    for (int j = 0; j < 8; ++j) {
        float v = racc[j];
        v += __shfl_xor(v, 1, 32);
        v += __shfl_xor(v, 2, 32);
        v += __shfl_xor(v, 4, 32);
        v += __shfl_xor(v, 8, 32);
        racc[j] = v;
    }

    // scale = sqrt(pi/2) / NUM_PROJ
    const float SCALE = 0.0048957584f;
    if (r == 0) {
        const int outBase = rowBase + wave * 16 + hi * 8;
        #pragma unroll
        for (int j = 0; j < 8; ++j)
            out[outBase + j] = SCALE * racc[j];
    }
}

extern "C" void kernel_launch(void* const* d_in, const int* in_sizes, int n_in,
                              void* d_out, int out_size, void* d_ws, size_t ws_size,
                              hipStream_t stream) {
    (void)n_in; (void)d_ws; (void)ws_size; (void)out_size;
    const float* query = (const float*)d_in[0];
    const float* key   = (const float*)d_in[1];
    const float* S     = (const float*)d_in[2];
    float*       out   = (float*)d_out;

    const int rows   = in_sizes[0] / DIM;          // B*H*SEQ = 131072
    const int blocks = rows / ROWS_PB;             // 1024

    qjl_sign_proj_kernel<<<blocks, THREADS, 0, stream>>>(query, key, S, out, rows);
}
